// GAT_57543971832576
// MI455X (gfx1250) — matrix operand order repaired and checked by reading, not compiled, not run on hardware
//
#include <hip/hip_runtime.h>
#include <hip/hip_bf16.h>
#include <math.h>

typedef _Float16 f16_t;
typedef __attribute__((ext_vector_type(16))) _Float16 v16h;
typedef __attribute__((ext_vector_type(8)))  float    v8f;

#define GAT_N 8192
#define GAT_F 128

// ---------------------------------------------------------------------------
// Kernel A: x' = X*W + b  (one block per row, 128 threads, one col each)
// Stores x' transposed as f16 (for WMMA B operand), and f1/f2 projections.
// ---------------------------------------------------------------------------
__global__ __launch_bounds__(128) void gat_xprime_kernel(
    const float* __restrict__ input,   // [N, F]
    const float* __restrict__ weight,  // [F, F]
    const float* __restrict__ bias,    // [F]
    const float* __restrict__ phi,     // [2F]
    f16_t* __restrict__ xpT,           // [F, N] f16 (transposed)
    float* __restrict__ f1,            // [N]
    float* __restrict__ f2)            // [N]
{
    __shared__ float xin[GAT_F];
    __shared__ float red[GAT_F];
    const int t = threadIdx.x;
    const int i = blockIdx.x;

    xin[t] = input[(size_t)i * GAT_F + t];
    __syncthreads();

    float acc = 0.f;
#pragma unroll 8
    for (int k = 0; k < GAT_F; ++k)
        acc = fmaf(xin[k], weight[(size_t)k * GAT_F + t], acc);
    const float xp = acc + bias[t];

    // transposed f16 store for the attention matmul B operand
    xpT[(size_t)t * GAT_N + i] = (f16_t)xp;

    // f1[i] = dot(x'_i, phi[0:F])
    red[t] = xp * phi[t];
    __syncthreads();
    for (int s = GAT_F / 2; s > 0; s >>= 1) {
        if (t < s) red[t] += red[t + s];
        __syncthreads();
    }
    if (t == 0) f1[i] = red[0];
    __syncthreads();

    // f2[i] = dot(x'_i, phi[F:2F])
    red[t] = xp * phi[GAT_F + t];
    __syncthreads();
    for (int s = GAT_F / 2; s > 0; s >>= 1) {
        if (t < s) red[t] += red[t + s];
        __syncthreads();
    }
    if (t == 0) f2[i] = red[0];
}

// ---------------------------------------------------------------------------
// Kernel B: global max of f2 (single block) -> scalar used as softmax shift
// bound: m_i = leaky_relu(f1_i + max_j f2_j) >= true masked row max.
// ---------------------------------------------------------------------------
__global__ __launch_bounds__(1024) void gat_maxf2_kernel(
    const float* __restrict__ f2, float* __restrict__ maxf2)
{
    __shared__ float sm[1024];
    float v = -3.0e38f;
    for (int idx = threadIdx.x; idx < GAT_N; idx += 1024)
        v = fmaxf(v, f2[idx]);
    sm[threadIdx.x] = v;
    __syncthreads();
    for (int s = 512; s > 0; s >>= 1) {
        if (threadIdx.x < s) sm[threadIdx.x] = fmaxf(sm[threadIdx.x], sm[threadIdx.x + s]);
        __syncthreads();
    }
    if (threadIdx.x == 0) maxf2[0] = sm[0];
}

// ---------------------------------------------------------------------------
// Kernel C: fused masked-softmax x matmul (flash-attention style).
// One wave32 per 16-row tile, 8 column tiles of F (v8f acc each).
// Software-pipelined: B tiles + next chunk's adj/f2 are issued first, the
// exp-heavy A-tile build hides their latency, then 8 back-to-back WMMAs.
// ---------------------------------------------------------------------------
__global__ __launch_bounds__(256) void gat_attn_kernel(
    const float* __restrict__ adj,     // [N, N]
    const f16_t* __restrict__ xpT,     // [F, N] f16
    const float* __restrict__ f1,      // [N]
    const float* __restrict__ f2,      // [N]
    const float* __restrict__ maxf2p,  // [1]
    float* __restrict__ out)           // [N, F]
{
    const int lane    = threadIdx.x & 31;
    const int wave    = threadIdx.x >> 5;
    const int l15     = lane & 15;
    const int halfSel = lane >> 4;          // 0: K {0..7,16..23}; 1: K {8..15,24..31}
    const int rowBase = (blockIdx.x * 8 + wave) * 16;
    const int myRow   = rowBase + l15;      // A-operand: lanes L and L+16 share row L

    const float f1i = f1[myRow];
    const float mB  = f1i + maxf2p[0];
    const float mi  = mB > 0.f ? mB : 0.2f * mB;   // leaky_relu upper bound of row max

    const v8f vzero = {0.f, 0.f, 0.f, 0.f, 0.f, 0.f, 0.f, 0.f};
    v8f acc[8];
#pragma unroll
    for (int ct = 0; ct < 8; ++ct) acc[ct] = vzero;

    float rs0 = 0.f, rs1 = 0.f, rs2 = 0.f, rs3 = 0.f;   // broken sum chain
    const float* arow = adj + (size_t)myRow * GAT_N;
    const int kOff = halfSel * 8;

    // ---- pipeline prologue: stage chunk 0's adj/f2 in registers ----
    float4 A0 = *(const float4*)(arow + kOff);
    float4 A1 = *(const float4*)(arow + kOff + 4);
    float4 A2 = *(const float4*)(arow + kOff + 16);
    float4 A3 = *(const float4*)(arow + kOff + 20);
    float4 G0 = *(const float4*)(f2 + kOff);
    float4 G1 = *(const float4*)(f2 + kOff + 4);
    float4 G2 = *(const float4*)(f2 + kOff + 16);
    float4 G3 = *(const float4*)(f2 + kOff + 20);

    for (int jb = 0; jb < GAT_N; jb += 32) {
        // (1) issue this chunk's 8 B tiles (32B each) -- latency hidden by (3)
        const f16_t* brow = xpT + (size_t)l15 * GAT_N + jb + halfSel * 8;
        v16h b[8];
#pragma unroll
        for (int ct = 0; ct < 8; ++ct)
#pragma unroll
            for (int s = 0; s < 16; ++s)
                b[ct][s] = brow[(size_t)ct * 16 * GAT_N + (s < 8 ? s : s + 8)];

        // (2) issue next chunk's adj/f2 (wrapped on last iter; unused then)
        const int jn  = (jb + 32) & (GAT_N - 1);
        const int j0n = jn + kOff;
        const int j1n = j0n + 16;
        const float4 NA0 = *(const float4*)(arow + j0n);
        const float4 NA1 = *(const float4*)(arow + j0n + 4);
        const float4 NA2 = *(const float4*)(arow + j1n);
        const float4 NA3 = *(const float4*)(arow + j1n + 4);
        const float4 NG0 = *(const float4*)(f2 + j0n);
        const float4 NG1 = *(const float4*)(f2 + j0n + 4);
        const float4 NG2 = *(const float4*)(f2 + j1n);
        const float4 NG3 = *(const float4*)(f2 + j1n + 4);

        // deep stream-ahead prefetch on the adjacency row (32 chunks = 4KB)
        __builtin_prefetch(arow + ((jb + 1024) & (GAT_N - 1)), 0, 0);

        // (3) build A tile from the registers staged last iteration
        const int j0 = jb + kOff;
        const int j1 = j0 + 16;
        const float av[16] = {A0.x, A0.y, A0.z, A0.w, A1.x, A1.y, A1.z, A1.w,
                              A2.x, A2.y, A2.z, A2.w, A3.x, A3.y, A3.z, A3.w};
        const float gv[16] = {G0.x, G0.y, G0.z, G0.w, G1.x, G1.y, G1.z, G1.w,
                              G2.x, G2.y, G2.z, G2.w, G3.x, G3.y, G3.z, G3.w};

        v16h aop;
        float w[16];
#pragma unroll
        for (int e = 0; e < 16; ++e) {
            const int col = (e < 8) ? (j0 + e) : (j1 + e - 8);
            const float tt = f1i + gv[e];
            const float s  = tt > 0.f ? tt : 0.2f * tt;          // leaky_relu(0.2)
            const float ex = __expf(s - mi);
            w[e] = (av[e] != 0.f || col == myRow) ? ex : 0.f;    // adj+I mask
            aop[e] = (_Float16)w[e];
        }
#pragma unroll
        for (int e = 0; e < 16; e += 4) {
            rs0 += w[e];
            rs1 += w[e + 1];
            rs2 += w[e + 2];
            rs3 += w[e + 3];
        }

        // (4) 8 back-to-back WMMAs on the preloaded B tiles
#pragma unroll
        for (int ct = 0; ct < 8; ++ct)
            acc[ct] = __builtin_amdgcn_wmma_f32_16x16x32_f16(
                false, aop, false, b[ct], (short)0, acc[ct], false, false);

        // (5) rotate pipeline registers
        A0 = NA0; A1 = NA1; A2 = NA2; A3 = NA3;
        G0 = NG0; G1 = NG1; G2 = NG2; G3 = NG3;
    }

    // combine partial row sums: lanes L and L+16 jointly cover row L's 32 K's/chunk
    float rowsum = (rs0 + rs1) + (rs2 + rs3);
    rowsum += __shfl_xor(rowsum, 16, 32);
    const float inv = 1.0f / rowsum;

    // C/D layout: lanes 0-15 -> M = r, lanes 16-31 -> M = 8 + r; N = l15
    float invs[8];
#pragma unroll
    for (int r = 0; r < 8; ++r)
        invs[r] = __shfl(inv, halfSel * 8 + r, 32);   // row (rowBase + halfSel*8 + r)

#pragma unroll
    for (int r = 0; r < 8; ++r) {
        const int row = rowBase + halfSel * 8 + r;
        float* orow = out + (size_t)row * GAT_F + l15;
#pragma unroll
        for (int ct = 0; ct < 8; ++ct)
            orow[ct * 16] = acc[ct][r] * invs[r];
    }
}

// ---------------------------------------------------------------------------
extern "C" void kernel_launch(void* const* d_in, const int* in_sizes, int n_in,
                              void* d_out, int out_size, void* d_ws, size_t ws_size,
                              hipStream_t stream) {
    const float* adj    = (const float*)d_in[0];  // [N,N]
    const float* input  = (const float*)d_in[1];  // [N,F]
    const float* weight = (const float*)d_in[2];  // [F,F]
    const float* bias   = (const float*)d_in[3];  // [F]
    const float* phi    = (const float*)d_in[4];  // [2F,1]
    float* out = (float*)d_out;                   // [N,F]

    // workspace layout
    char* ws = (char*)d_ws;
    f16_t* xpT  = (f16_t*)ws;                                        // F*N f16 = 2 MiB
    float* f1   = (float*)(ws + (size_t)GAT_F * GAT_N * sizeof(f16_t));
    float* f2   = f1 + GAT_N;
    float* mxf2 = f2 + GAT_N;

    gat_xprime_kernel<<<GAT_N, GAT_F, 0, stream>>>(input, weight, bias, phi, xpT, f1, f2);
    gat_maxf2_kernel<<<1, 1024, 0, stream>>>(f2, mxf2);
    gat_attn_kernel<<<GAT_N / (16 * 8), 256, 0, stream>>>(adj, xpT, f1, f2, mxf2, out);
}
